// NMPN_8340826489581
// MI455X (gfx1250) — matrix operand order repaired and checked by reading, not compiled, not run on hardware
//
#include <hip/hip_runtime.h>
#include <cstdint>

#define N_ATOMS   100000
#define N_BONDS   200000
#define MAX_NB    6
#define HIDDEN    256
#define DEPTH     6
#define ATOM_FDIM 39
#define BOND_FDIM 11
#define W_COLS    (HIDDEN + BOND_FDIM)   // 267

typedef __attribute__((ext_vector_type(16))) __bf16 v16bf;
typedef __attribute__((ext_vector_type(8)))  float  v8f;

// float -> bf16 with round-to-nearest-even (returned in low 16 bits)
static __device__ __forceinline__ uint32_t f2bf(float f) {
    union { float f; uint32_t u; } x; x.f = f;
    uint32_t u = x.u;
    return ((u + 0x7FFFu + ((u >> 16) & 1u)) >> 16) & 0xFFFFu;
}

// ---------------------------------------------------------------------------
// src[a*6+j] = (b==0) ? -1 : all_bonds[b][1],  b = aoutgraph[a][j]
// ---------------------------------------------------------------------------
__global__ __launch_bounds__(256) void mpnn_prep_src(
    const int* __restrict__ aog, const int* __restrict__ ab, int* __restrict__ src)
{
    int i = blockIdx.x * 256 + threadIdx.x;
    if (i >= N_ATOMS * MAX_NB) return;
    int b = aog[i];
    src[i] = (b == 0) ? -1 : ab[b * 2 + 1];
}

// ---------------------------------------------------------------------------
// Pack W_node[:, :256] into per-lane WMMA B-fragment order (bf16), contiguous k matching A.
// Layout index: (((ntile*8 + ktile)*32 + lane)*8 + v) dwords.
//   n = ntile*16 + (lane&15);  k = ktile*32 + (lane>=16 ? 16:0) + 2v, 2v+1
//   B[k][n] = W_node[n][k]
// ---------------------------------------------------------------------------
__global__ __launch_bounds__(256) void mpnn_prep_w(
    const float* __restrict__ W_node, uint32_t* __restrict__ Wpack)
{
    int i = blockIdx.x * 256 + threadIdx.x;      // 0 .. 32767
    if (i >= 16 * 8 * 32 * 8) return;
    int v    = i & 7;
    int lane = (i >> 3) & 31;
    int kt   = (i >> 8) & 7;
    int nt   = i >> 11;
    int n  = nt * 16 + (lane & 15);
    int k0 = kt * 32 + ((lane >> 4) * 16) + 2 * v;
    const float* wr = W_node + (size_t)n * W_COLS + k0;
    Wpack[i] = f2bf(wr[0]) | (f2bf(wr[1]) << 16);
}

// ---------------------------------------------------------------------------
// bias[a][h] = relu(fatoms[a]·W_nin[h]) + fb_sum[a]·W_node[h][256:267]
// Hn0[a][h]  = relu(fatoms[a]·W_nin[h])
// One block per atom, one thread per hidden unit.
// ---------------------------------------------------------------------------
__global__ __launch_bounds__(256) void mpnn_prep_bias(
    const float* __restrict__ fatoms, const float* __restrict__ fbonds,
    const int* __restrict__ aog,
    const float* __restrict__ W_nin, const float* __restrict__ W_node,
    float* __restrict__ bias, float* __restrict__ Hn0)
{
    __shared__ float fa[ATOM_FDIM];
    __shared__ float fs[BOND_FDIM];
    const int a = blockIdx.x;
    const int t = threadIdx.x;

    if (t < ATOM_FDIM) fa[t] = fatoms[(size_t)a * ATOM_FDIM + t];
    if (t < BOND_FDIM) {
        float s = 0.f;
        #pragma unroll
        for (int j = 0; j < MAX_NB; j++) {
            int b = aog[a * MAX_NB + j];
            s += fbonds[(size_t)b * BOND_FDIM + t];
        }
        fs[t] = s;
    }
    __syncthreads();

    float h0 = 0.f;
    const float* wr = W_nin + (size_t)t * ATOM_FDIM;
    #pragma unroll
    for (int k = 0; k < ATOM_FDIM; k++) h0 += wr[k] * fa[k];
    h0 = fmaxf(h0, 0.f);

    float fb = 0.f;
    const float* wn = W_node + (size_t)t * W_COLS + HIDDEN;
    #pragma unroll
    for (int k = 0; k < BOND_FDIM; k++) fb += wn[k] * fs[k];

    bias[(size_t)a * HIDDEN + t] = h0 + fb;
    Hn0 [(size_t)a * HIDDEN + t] = h0;
}

// ---------------------------------------------------------------------------
// One iteration: per block, 16 atoms.
//   Phase 1: agg[m][:] = sum_j Hn_in[src[m][j]]   (f32 accumulated in regs,
//            packed to bf16 row-major in LDS, row stride 129 dwords)
//   Phase 2: 8 waves; wave w computes N-tiles 2w, 2w+1 with bf16 WMMA,
//            C preloaded from bias, 8 K-steps; relu; store.
// ---------------------------------------------------------------------------
__global__ __launch_bounds__(256) void mpnn_iter(
    const float* __restrict__ Hn_in, const float* __restrict__ bias,
    const int* __restrict__ src, const uint32_t* __restrict__ Wpack,
    float* __restrict__ Hn_out, float* __restrict__ final_out)
{
    __shared__ uint32_t aggbf[16 * 129];       // 16 rows x 256 bf16, +1 dword pad
    const int t     = threadIdx.x;
    const int atile = blockIdx.x;

    // ---- Phase 1: gather-sum into LDS (bf16) -------------------------------
    {
        const int m   = t >> 4;                // atom row within tile
        const int seg = t & 15;                // 16-dim segment
        const int a   = atile * 16 + m;
        float acc[16];
        #pragma unroll
        for (int i = 0; i < 16; i++) acc[i] = 0.f;
        const int* sp = src + a * MAX_NB;
        #pragma unroll
        for (int j = 0; j < MAX_NB; j++) {
            int s = sp[j];
            if (s >= 0) {
                const float4* p = (const float4*)(Hn_in + (size_t)s * HIDDEN + seg * 16);
                #pragma unroll
                for (int q = 0; q < 4; q++) {
                    float4 r = p[q];
                    acc[q * 4 + 0] += r.x; acc[q * 4 + 1] += r.y;
                    acc[q * 4 + 2] += r.z; acc[q * 4 + 3] += r.w;
                }
            }
        }
        uint32_t* row = &aggbf[m * 129];
        #pragma unroll
        for (int i = 0; i < 8; i++)
            row[seg * 8 + i] = f2bf(acc[2 * i]) | (f2bf(acc[2 * i + 1]) << 16);
    }
    __syncthreads();

    // ---- Phase 2: WMMA -----------------------------------------------------
    const int w    = t >> 5;
    const int lane = t & 31;
    const int ncol = lane & 15;                // D: N = lane (mod 16)
    const int mrow = (lane >> 4) << 3;         // D: M base 0 or 8
    const int n0 = w * 2, n1 = n0 + 1;         // two 16-wide N tiles per wave

    v8f c0, c1;
    const float* bb = bias + (size_t)atile * 16 * HIDDEN;
    #pragma unroll
    for (int r = 0; r < 8; r++) {
        c0[r] = bb[(mrow + r) * HIDDEN + n0 * 16 + ncol];
        c1[r] = bb[(mrow + r) * HIDDEN + n1 * 16 + ncol];
    }

    const int am    = lane & 15;               // A: M = lane (mod 16)
    const int khalf = (lane >> 4) * 16;        // A: K-half select
    const uint32_t* arow = &aggbf[am * 129];

    #pragma unroll
    for (int kt = 0; kt < 8; kt++) {
        union { v16bf v; uint32_t u[8]; } A, B0, B1;
        #pragma unroll
        for (int v = 0; v < 8; v++) {
            int k = kt * 32 + khalf + 2 * v;
            A.u[v] = arow[k >> 1];
        }
        const uint4* p0 = (const uint4*)(Wpack + ((size_t)(n0 * 8 + kt) * 32 + lane) * 8);
        uint4 q0 = p0[0], q1 = p0[1];
        B0.u[0] = q0.x; B0.u[1] = q0.y; B0.u[2] = q0.z; B0.u[3] = q0.w;
        B0.u[4] = q1.x; B0.u[5] = q1.y; B0.u[6] = q1.z; B0.u[7] = q1.w;
        const uint4* p1 = (const uint4*)(Wpack + ((size_t)(n1 * 8 + kt) * 32 + lane) * 8);
        uint4 r0 = p1[0], r1 = p1[1];
        B1.u[0] = r0.x; B1.u[1] = r0.y; B1.u[2] = r0.z; B1.u[3] = r0.w;
        B1.u[4] = r1.x; B1.u[5] = r1.y; B1.u[6] = r1.z; B1.u[7] = r1.w;

        c0 = __builtin_amdgcn_wmma_f32_16x16x32_bf16(false, A.v, false, B0.v,
                                                     (short)0, c0, false, false);
        c1 = __builtin_amdgcn_wmma_f32_16x16x32_bf16(false, A.v, false, B1.v,
                                                     (short)0, c1, false, false);
    }

    // ---- relu + store ------------------------------------------------------
    float o0[8], o1[8];
    #pragma unroll
    for (int r = 0; r < 8; r++) {
        o0[r] = fmaxf(c0[r], 0.f);
        o1[r] = fmaxf(c1[r], 0.f);
    }
    const int mg0 = atile * 16 + mrow;
    #pragma unroll
    for (int r = 0; r < 8; r++) {
        Hn_out[(size_t)(mg0 + r) * HIDDEN + n0 * 16 + ncol] = o0[r];
        Hn_out[(size_t)(mg0 + r) * HIDDEN + n1 * 16 + ncol] = o1[r];
    }
    if (final_out) {   // output is Hn.T : out[h][a], per-lane contiguous in a
        float4* f0 = (float4*)(final_out + (size_t)(n0 * 16 + ncol) * N_ATOMS + mg0);
        f0[0] = make_float4(o0[0], o0[1], o0[2], o0[3]);
        f0[1] = make_float4(o0[4], o0[5], o0[6], o0[7]);
        float4* f1 = (float4*)(final_out + (size_t)(n1 * 16 + ncol) * N_ATOMS + mg0);
        f1[0] = make_float4(o1[0], o1[1], o1[2], o1[3]);
        f1[1] = make_float4(o1[4], o1[5], o1[6], o1[7]);
    }
}

// ---------------------------------------------------------------------------
extern "C" void kernel_launch(void* const* d_in, const int* in_sizes, int n_in,
                              void* d_out, int out_size, void* d_ws, size_t ws_size,
                              hipStream_t stream)
{
    const float* fatoms    = (const float*)d_in[0];
    const float* fbonds    = (const float*)d_in[1];
    const int*   aoutgraph = (const int*)  d_in[2];
    const int*   all_bonds = (const int*)  d_in[3];
    const float* W_nin     = (const float*)d_in[4];
    const float* W_node    = (const float*)d_in[5];
    (void)in_sizes; (void)n_in; (void)out_size; (void)ws_size;

    char* ws = (char*)d_ws;
    size_t off = 0;
    auto alloc = [&](size_t bytes) -> void* {
        void* p = ws + off;
        off = (off + bytes + 255) & ~(size_t)255;
        return p;
    };
    int*      src   = (int*)     alloc((size_t)N_ATOMS * MAX_NB * sizeof(int));
    uint32_t* Wpack = (uint32_t*)alloc((size_t)16 * 8 * 32 * 8 * sizeof(uint32_t));
    float*    bias  = (float*)   alloc((size_t)N_ATOMS * HIDDEN * sizeof(float));
    float*    HnA   = (float*)   alloc((size_t)N_ATOMS * HIDDEN * sizeof(float));
    float*    HnB   = (float*)   alloc((size_t)N_ATOMS * HIDDEN * sizeof(float));

    mpnn_prep_src <<<(N_ATOMS * MAX_NB + 255) / 256, 256, 0, stream>>>(aoutgraph, all_bonds, src);
    mpnn_prep_w   <<<(16 * 8 * 32 * 8 + 255) / 256, 256, 0, stream>>>(W_node, Wpack);
    mpnn_prep_bias<<<N_ATOMS, 256, 0, stream>>>(fatoms, fbonds, aoutgraph, W_nin, W_node, bias, HnA);

    const float* cur = HnA;   // HnA holds h0
    float*       nxt = HnB;
    for (int it = 0; it < DEPTH; it++) {
        float* fin = (it == DEPTH - 1) ? (float*)d_out : nullptr;
        mpnn_iter<<<N_ATOMS / 16, 256, 0, stream>>>(cur, bias, src, Wpack, nxt, fin);
        float* tmp = nxt;
        nxt = (float*)cur;
        cur = tmp;
    }
}
